// SwinTransformerV2CrBlock_8821862826387
// MI455X (gfx1250) — matrix operand order repaired and checked by reading, not compiled, not run on hardware
//
#include <hip/hip_runtime.h>
#include <hip/hip_bf16.h>

// ---------------------------------------------------------------------------
// SwinTransformerV2 CR block for gfx1250 (MI455X), wave32 + WMMA bf16.
// B=16, H=W=64, C=256, NH=8, hd=32, WS=8, SS=4, L=64, NW=64, Bw=1024, HID=1024
// ---------------------------------------------------------------------------

typedef __attribute__((ext_vector_type(16))) __bf16 v16bf;
typedef __attribute__((ext_vector_type(8)))  float  v8f;

#if defined(__has_builtin)
#if __has_builtin(__builtin_amdgcn_sched_barrier)
#define SCHED_FENCE() __builtin_amdgcn_sched_barrier(0)
#endif
#endif
#ifndef SCHED_FENCE
#define SCHED_FENCE()
#endif

union Frag {
    v16bf v;
    uint4 q[2];
};

__device__ __forceinline__ unsigned short f2bf(float f) {
    unsigned u = __float_as_uint(f);
    if ((u & 0x7fffffffu) > 0x7f800000u) return (unsigned short)0x7fc0;
    u += 0x7fffu + ((u >> 16) & 1u);
    return (unsigned short)(u >> 16);
}

__device__ __forceinline__ unsigned pack2bf(float lo, float hi) {
    return (unsigned)f2bf(lo) | ((unsigned)f2bf(hi) << 16);
}

// A fragment (16x32 bf16, M x K), A row-major with leading dim lda (elements).
// ISA layout: lane<16 -> row m_base+lane, K = kb+{0..7,16..23};
//             lane>=16 -> row m_base+lane-16, K = kb+{8..15,24..31}.
__device__ __forceinline__ Frag load_a(const unsigned short* A, int lda,
                                       int m_base, int k_base, int lane) {
    Frag f;
    int m = m_base + (lane & 15);
    int k = k_base + ((lane & 16) ? 8 : 0);
    f.q[0] = *(const uint4*)(A + m * lda + k);
    f.q[1] = *(const uint4*)(A + m * lda + k + 16);
    return f;
}

// B fragment (32x16 bf16, K x N), B row-major with leading dim ldb (elements).
// ISA layout: lane<16 -> col n_base+lane, K = kb+{0..7,16..23};
//             lane>=16 -> col n_base+lane-16, K = kb+{8..15,24..31}.
__device__ __forceinline__ Frag load_b(const unsigned short* B, int ldb,
                                       int k_base, int n_base, int lane) {
    Frag f;
    const unsigned short* p = B + (size_t)(k_base + ((lane & 16) ? 8 : 0)) * ldb + n_base + (lane & 15);
    unsigned w[8];
#pragma unroll
    for (int i = 0; i < 8; ++i) {
        int kr = 2 * i + ((i >= 4) ? 8 : 0);
        w[i] = (unsigned)p[(size_t)kr * ldb] | ((unsigned)p[(size_t)(kr + 1) * ldb] << 16);
    }
    f.q[0] = make_uint4(w[0], w[1], w[2], w[3]);
    f.q[1] = make_uint4(w[4], w[5], w[6], w[7]);
    return f;
}

__device__ __forceinline__ v8f wmma_bf16(const Frag& a, const Frag& b, v8f c) {
    return __builtin_amdgcn_wmma_f32_16x16x32_bf16(false, a.v, false, b.v,
                                                   (short)0, c, false, false);
}

// shifted-window region id for token t inside window (wh, ww)
__device__ __forceinline__ int cnt_tok(int t, int wh, int ww) {
    int r = t >> 3, c = t & 7;
    int rh = (wh == 7) ? (r < 4 ? 1 : 2) : 0;
    int rw = (ww == 7) ? (c < 4 ? 1 : 2) : 0;
    return rh * 3 + rw;
}

// ---------------------------------------------------------------------------
// K0: weight transpose + fp32->bf16  (wT[k][n] = w[n][k])
// ---------------------------------------------------------------------------
__global__ __launch_bounds__(256) void k_prep(
    const float* __restrict__ qkv_w, const float* __restrict__ proj_w,
    const float* __restrict__ w1,    const float* __restrict__ w2,
    unsigned short* __restrict__ wqkvT, unsigned short* __restrict__ wprojT,
    unsigned short* __restrict__ w1T,   unsigned short* __restrict__ w2T) {
    int i = blockIdx.x * 256 + threadIdx.x;
    if (i < 196608) {                        // 256 x 768
        int k = i / 768, n = i % 768;
        wqkvT[i] = f2bf(qkv_w[n * 256 + k]);
    } else if (i < 262144) {                 // 256 x 256
        int j = i - 196608; int k = j / 256, n = j % 256;
        wprojT[j] = f2bf(proj_w[n * 256 + k]);
    } else if (i < 524288) {                 // 256 x 1024
        int j = i - 262144; int k = j / 1024, n = j % 1024;
        w1T[j] = f2bf(w1[n * 256 + k]);
    } else {                                 // 1024 x 256
        int j = i - 524288; int k = j / 256, n = j % 256;
        w2T[j] = f2bf(w2[n * 1024 + k]);
    }
}

// ---------------------------------------------------------------------------
// K0b: log-CPB relative-position bias table via meta MLP  -> biasT[8][64][64]
// ---------------------------------------------------------------------------
__global__ __launch_bounds__(256) void k_bias(
    const float* __restrict__ mw1, const float* __restrict__ mb1,
    const float* __restrict__ mw2, const float* __restrict__ mb2,
    float* __restrict__ biasT) {
    int idx = blockIdx.x * 256 + threadIdx.x;   // 0..4095 = i*64+j
    int i = idx >> 6, j = idx & 63;
    float d0 = (float)((i >> 3) - (j >> 3));
    float d1 = (float)((i & 7) - (j & 7));
    float s0 = (d0 > 0.f) ? 1.f : ((d0 < 0.f) ? -1.f : 0.f);
    float s1 = (d1 > 0.f) ? 1.f : ((d1 < 0.f) ? -1.f : 0.f);
    d0 = s0 * log1pf(fabsf(d0));
    d1 = s1 * log1pf(fabsf(d1));
    float o[8];
#pragma unroll
    for (int h = 0; h < 8; ++h) o[h] = 0.f;
    for (int h = 0; h < 384; ++h) {
        float v = fmaxf(0.f, d0 * mw1[h * 2] + d1 * mw1[h * 2 + 1] + mb1[h]);
#pragma unroll
        for (int nh = 0; nh < 8; ++nh) o[nh] += v * mw2[nh * 384 + h];
    }
#pragma unroll
    for (int nh = 0; nh < 8; ++nh) biasT[nh * 4096 + idx] = o[nh] + mb2[nh];
}

// ---------------------------------------------------------------------------
// K1: shift + window gather + QKV GEMM (64x768, K=256) -> q,k,v fp32
//     layout [Bw][NH][L][hd]
// ---------------------------------------------------------------------------
__global__ __launch_bounds__(256) void k_qkv(
    const float* __restrict__ x, const unsigned short* __restrict__ wqkvT,
    const float* __restrict__ qkv_b,
    float* __restrict__ qf, float* __restrict__ kf, float* __restrict__ vf) {
    __shared__ __align__(16) unsigned short Ab[64 * 256];
    int w = blockIdx.x;
    int b = w >> 6, widx = w & 63, wh = widx >> 3, ww = widx & 7;
    int tid = threadIdx.x;
    {   // gather window tokens (shifted) -> bf16 LDS A tile
        int t = tid >> 2, qc = tid & 3;
        int r = t >> 3, c = t & 7;
        int hh = (wh * 8 + r + 4) & 63;
        int wc = (ww * 8 + c + 4) & 63;
        const float4* src =
            (const float4*)(x + ((size_t)(b * 4096 + hh * 64 + wc)) * 256 + qc * 64);
        unsigned* dst = (unsigned*)(Ab + t * 256 + qc * 64);
#pragma unroll
        for (int i = 0; i < 16; ++i) {
            float4 v4 = src[i];
            dst[2 * i]     = pack2bf(v4.x, v4.y);
            dst[2 * i + 1] = pack2bf(v4.z, v4.w);
        }
    }
    __syncthreads();
    int wv = tid >> 5, lane = tid & 31;
    int m_base = (wv & 3) * 16;
    int nhalf = wv >> 2;
    int col = lane & 15;
    int r0 = m_base + ((lane & 16) ? 8 : 0);
    for (int ni = 0; ni < 24; ++ni) {
        int n = nhalf * 384 + ni * 16;
        // preload ALL B fragments; sched barrier keeps them hoisted as one
        // load clause so the WMMA chain only takes progressive waits.
        Frag bf8[8];
#pragma unroll
        for (int kk = 0; kk < 8; ++kk)
            bf8[kk] = load_b(wqkvT, 768, kk * 32, n, lane);
        SCHED_FENCE();
        v8f acc = {};
#pragma unroll
        for (int kk = 0; kk < 8; ++kk) {
            Frag a = load_a(Ab, 256, m_base, kk * 32, lane);
            acc = wmma_bf16(a, bf8[kk], acc);
        }
        SCHED_FENCE();
        int gn = n + col;
        float bias = qkv_b[gn];
        int third = gn >> 8, cc = gn & 255, head = cc >> 5, hdi = cc & 31;
        float* dst = (third == 0) ? qf : ((third == 1) ? kf : vf);
        size_t base = (((size_t)w * 8 + head) * 64) * 32 + hdi;
#pragma unroll
        for (int i = 0; i < 8; ++i)
            dst[base + (size_t)(r0 + i) * 32] = acc[i] + bias;
    }
}

// ---------------------------------------------------------------------------
// K2: attention per (window, head): cosine-normalize -> S -> softmax -> P@V
//     128 threads = 4 waves = 4 heads; grid = Bw*2
// ---------------------------------------------------------------------------
__global__ __launch_bounds__(128) void k_attn(
    const float* __restrict__ qf, const float* __restrict__ kf,
    const float* __restrict__ vf, const float* __restrict__ logit_scale,
    const float* __restrict__ biasT, unsigned short* __restrict__ ctx) {
    // per-wave region: qn[64][32] | knT[32][64] | v[64][32] | sP[64][64]
    __shared__ __align__(16) unsigned short smem[4 * 10240];
    int w = blockIdx.x >> 1;
    int widx = w & 63, wh = widx >> 3, ww = widx & 7;
    int tid = threadIdx.x, wv = tid >> 5, lane = tid & 31;
    int h = ((blockIdx.x & 1) << 2) + wv;
    unsigned short* qn  = smem + wv * 10240;
    unsigned short* knT = qn + 2048;
    unsigned short* vb  = qn + 4096;
    unsigned short* sP  = qn + 6144;

    size_t base = (((size_t)w * 8 + h) * 64) * 32;
    const float* qp = qf + base;
    const float* kp = kf + base;
    const float* vp = vf + base;

    // fp32 cosine normalization -> bf16 tiles (k pre-transposed for B frags)
    for (int r = 0; r < 64; ++r) {
        float qv = qp[r * 32 + lane];
        float ss = qv * qv;
#pragma unroll
        for (int m = 16; m; m >>= 1) ss += __shfl_xor(ss, m, 32);
        qn[r * 32 + lane] = f2bf(qv / fmaxf(sqrtf(ss), 1e-12f));

        float kv = kp[r * 32 + lane];
        float ks = kv * kv;
#pragma unroll
        for (int m = 16; m; m >>= 1) ks += __shfl_xor(ks, m, 32);
        knT[lane * 64 + r] = f2bf(kv / fmaxf(sqrtf(ks), 1e-12f));

        vb[r * 32 + lane] = f2bf(vp[r * 32 + lane]);
    }
    __syncthreads();

    float sc = expf(fminf(logit_scale[h], 4.60517018598809136804f)); // ln(100)
    int col = lane & 15;
    int hib = (lane & 16) ? 8 : 0;

    // S = qn @ knT, scale + CPB bias + analytic shift mask, softmax, -> sP bf16
    for (int mt = 0; mt < 4; ++mt) {
        v8f acc[4];
#pragma unroll
        for (int nt = 0; nt < 4; ++nt) {
            Frag a  = load_a(qn, 32, mt * 16, 0, lane);
            Frag bf = load_b(knT, 64, 0, nt * 16, lane);
            v8f z = {};
            acc[nt] = wmma_bf16(a, bf, z);
        }
#pragma unroll
        for (int i = 0; i < 8; ++i) {
            int row = mt * 16 + i + hib;
            int cr = cnt_tok(row, wh, ww);
            float l[4];
            float mx = -1e30f;
#pragma unroll
            for (int nt = 0; nt < 4; ++nt) {
                int cj = nt * 16 + col;
                float lg = acc[nt][i] * sc + biasT[h * 4096 + row * 64 + cj];
                if (cr != cnt_tok(cj, wh, ww)) lg -= 100.f;
                l[nt] = lg;
                mx = fmaxf(mx, lg);
            }
#pragma unroll
            for (int m = 1; m < 16; m <<= 1) mx = fmaxf(mx, __shfl_xor(mx, m, 32));
            float s = 0.f;
#pragma unroll
            for (int nt = 0; nt < 4; ++nt) { l[nt] = expf(l[nt] - mx); s += l[nt]; }
#pragma unroll
            for (int m = 1; m < 16; m <<= 1) s += __shfl_xor(s, m, 32);
            float inv = 1.f / s;
#pragma unroll
            for (int nt = 0; nt < 4; ++nt)
                sP[row * 64 + nt * 16 + col] = f2bf(l[nt] * inv);
        }
    }
    __syncthreads();

    // ctx = P @ V  (64x32), head-concat bf16 output [w][t][h*32+d]
    for (int mt = 0; mt < 4; ++mt) {
#pragma unroll
        for (int nt = 0; nt < 2; ++nt) {
            v8f acc = {};
#pragma unroll
            for (int ks = 0; ks < 2; ++ks) {
                Frag a  = load_a(sP, 64, mt * 16, ks * 32, lane);
                Frag bf = load_b(vb, 32, ks * 32, nt * 16, lane);
                acc = wmma_bf16(a, bf, acc);
            }
#pragma unroll
            for (int i = 0; i < 8; ++i) {
                int row = mt * 16 + i + hib;
                ctx[((size_t)w * 64 + row) * 256 + h * 32 + nt * 16 + col] =
                    f2bf(acc[i]);
            }
        }
    }
}

// ---------------------------------------------------------------------------
// K3: proj GEMM (64x256, K=256) + LayerNorm1 + window-reverse/unshift scatter
//     + residual:  x1 = x + LN(o)   (x1 written into d_out)
// ---------------------------------------------------------------------------
__global__ __launch_bounds__(256) void k_proj(
    const unsigned short* __restrict__ ctx, const unsigned short* __restrict__ wprojT,
    const float* __restrict__ proj_b, const float* __restrict__ x,
    const float* __restrict__ g1, const float* __restrict__ b1n,
    float* __restrict__ x1) {
    __shared__ __align__(16) unsigned short Ab[64 * 256];
    __shared__ __align__(16) float Ob[64 * 256];
    int w = blockIdx.x;
    int b = w >> 6, widx = w & 63, wh = widx >> 3, ww = widx & 7;
    int tid = threadIdx.x;
    {   // copy bf16 ctx tile into LDS
        const uint4* src = (const uint4*)(ctx + (size_t)w * 64 * 256);
        uint4* dst = (uint4*)Ab;
        for (int i = tid; i < 2048; i += 256) dst[i] = src[i];
    }
    __syncthreads();
    int wv = tid >> 5, lane = tid & 31;
    int m_base = (wv & 3) * 16, nh2 = wv >> 2;
    int col = lane & 15;
    int r0 = m_base + ((lane & 16) ? 8 : 0);
    for (int ni = 0; ni < 8; ++ni) {
        int n = nh2 * 128 + ni * 16;
        Frag bf8[8];
#pragma unroll
        for (int kk = 0; kk < 8; ++kk)
            bf8[kk] = load_b(wprojT, 256, kk * 32, n, lane);
        SCHED_FENCE();
        v8f acc = {};
#pragma unroll
        for (int kk = 0; kk < 8; ++kk) {
            Frag a = load_a(Ab, 256, m_base, kk * 32, lane);
            acc = wmma_bf16(a, bf8[kk], acc);
        }
        SCHED_FENCE();
        float pb = proj_b[n + col];
#pragma unroll
        for (int i = 0; i < 8; ++i) Ob[(r0 + i) * 256 + n + col] = acc[i] + pb;
    }
    __syncthreads();
    // LayerNorm per row (4 threads per token) + scatter + residual
    int t = tid >> 2, part = tid & 3;
    float s = 0.f, sq = 0.f;
    const float4* row4 = (const float4*)(Ob + t * 256 + part * 64);
#pragma unroll
    for (int i = 0; i < 16; ++i) {
        float4 v4 = row4[i];
        s  += v4.x + v4.y + v4.z + v4.w;
        sq += v4.x * v4.x + v4.y * v4.y + v4.z * v4.z + v4.w * v4.w;
    }
    s  += __shfl_xor(s, 1, 32);  s  += __shfl_xor(s, 2, 32);
    sq += __shfl_xor(sq, 1, 32); sq += __shfl_xor(sq, 2, 32);
    float mean = s * (1.f / 256.f);
    float var  = sq * (1.f / 256.f) - mean * mean;
    float rstd = rsqrtf(var + 1e-5f);
    int r = t >> 3, c = t & 7;
    int hh = (wh * 8 + r + 4) & 63;
    int wc = (ww * 8 + c + 4) & 63;
    size_t drow = ((size_t)(b * 4096 + hh * 64 + wc)) * 256;
    for (int j = 0; j < 64; ++j) {
        int chn = part * 64 + j;
        float o = Ob[t * 256 + chn];
        x1[drow + chn] = x[drow + chn] + (o - mean) * rstd * g1[chn] + b1n[chn];
    }
}

// ---------------------------------------------------------------------------
// K4: fused MLP: h = GELU(x1 @ w1T + b1); o = h @ w2T + b2;
//     out = x1 + LN2(o).   Hidden (64x1024) chunked by 128, never hits HBM.
// ---------------------------------------------------------------------------
__global__ __launch_bounds__(256) void k_mlp(
    float* __restrict__ xio, const unsigned short* __restrict__ w1T,
    const float* __restrict__ b1m, const unsigned short* __restrict__ w2T,
    const float* __restrict__ b2m, const float* __restrict__ g2,
    const float* __restrict__ b2ln) {
    __shared__ __align__(16) unsigned short A1[64 * 256];
    __shared__ __align__(16) unsigned short Hb[64 * 128];
    __shared__ __align__(16) float H2[64 * 256];
    int g = blockIdx.x;
    int tid = threadIdx.x;
    size_t rowBase = (size_t)g * 64;
    {   // x1 tile fp32 -> bf16 LDS
        int t = tid >> 2, part = tid & 3;
        const float4* src = (const float4*)(xio + (rowBase + t) * 256 + part * 64);
        unsigned* dst = (unsigned*)(A1 + t * 256 + part * 64);
#pragma unroll
        for (int i = 0; i < 16; ++i) {
            float4 v4 = src[i];
            dst[2 * i]     = pack2bf(v4.x, v4.y);
            dst[2 * i + 1] = pack2bf(v4.z, v4.w);
        }
    }
    __syncthreads();
    int wv = tid >> 5, lane = tid & 31;
    int m_base = (wv & 3) * 16, nh2 = wv >> 2;
    int col = lane & 15;
    int r0 = m_base + ((lane & 16) ? 8 : 0);
    v8f accO[8];
#pragma unroll
    for (int i = 0; i < 8; ++i) accO[i] = (v8f){};
    for (int chk = 0; chk < 8; ++chk) {
        // GEMM1: hidden chunk [chk*128, chk*128+128), K=256, + GELU -> Hb bf16
#pragma unroll
        for (int ni = 0; ni < 4; ++ni) {
            int nl = nh2 * 64 + ni * 16;       // local col in chunk
            int ng = chk * 128 + nl;           // global hidden col
            Frag bf8[8];
#pragma unroll
            for (int kk = 0; kk < 8; ++kk)
                bf8[kk] = load_b(w1T, 1024, kk * 32, ng, lane);
            SCHED_FENCE();
            v8f acc = {};
#pragma unroll
            for (int kk = 0; kk < 8; ++kk) {
                Frag a = load_a(A1, 256, m_base, kk * 32, lane);
                acc = wmma_bf16(a, bf8[kk], acc);
            }
            SCHED_FENCE();
            float bb = b1m[ng + col];
#pragma unroll
            for (int i = 0; i < 8; ++i) {
                float hv = acc[i] + bb;
                float ge = 0.5f * hv * (1.f + erff(hv * 0.70710678118654752f));
                Hb[(r0 + i) * 128 + nl + col] = f2bf(ge);
            }
        }
        __syncthreads();
        // GEMM2 partial: accumulate out (64x256) over this chunk's K=128
#pragma unroll
        for (int ni = 0; ni < 8; ++ni) {
            int n = nh2 * 128 + ni * 16;
            Frag bf4[4];
#pragma unroll
            for (int kk = 0; kk < 4; ++kk)
                bf4[kk] = load_b(w2T, 256, chk * 128 + kk * 32, n, lane);
            SCHED_FENCE();
#pragma unroll
            for (int kk = 0; kk < 4; ++kk) {
                Frag a = load_a(Hb, 128, m_base, kk * 32, lane);
                accO[ni] = wmma_bf16(a, bf4[kk], accO[ni]);
            }
            SCHED_FENCE();
        }
        __syncthreads();
    }
    // bias2 -> LDS fp32 for row-wise LayerNorm
#pragma unroll
    for (int ni = 0; ni < 8; ++ni) {
        int n = nh2 * 128 + ni * 16;
        float bb = b2m[n + col];
#pragma unroll
        for (int i = 0; i < 8; ++i) H2[(r0 + i) * 256 + n + col] = accO[ni][i] + bb;
    }
    __syncthreads();
    // LN2 + residual (in-place on xio)
    int t = tid >> 2, part = tid & 3;
    float s = 0.f, sq = 0.f;
    const float4* row4 = (const float4*)(H2 + t * 256 + part * 64);
#pragma unroll
    for (int i = 0; i < 16; ++i) {
        float4 v4 = row4[i];
        s  += v4.x + v4.y + v4.z + v4.w;
        sq += v4.x * v4.x + v4.y * v4.y + v4.z * v4.z + v4.w * v4.w;
    }
    s  += __shfl_xor(s, 1, 32);  s  += __shfl_xor(s, 2, 32);
    sq += __shfl_xor(sq, 1, 32); sq += __shfl_xor(sq, 2, 32);
    float mean = s * (1.f / 256.f);
    float var  = sq * (1.f / 256.f) - mean * mean;
    float rstd = rsqrtf(var + 1e-5f);
    for (int j = 0; j < 64; ++j) {
        int chn = part * 64 + j;
        size_t idx = (rowBase + t) * 256 + chn;
        float o = H2[t * 256 + chn];
        xio[idx] = xio[idx] + (o - mean) * rstd * g2[chn] + b2ln[chn];
    }
}

// ---------------------------------------------------------------------------
extern "C" void kernel_launch(void* const* d_in, const int* in_sizes, int n_in,
                              void* d_out, int out_size, void* d_ws, size_t ws_size,
                              hipStream_t stream) {
    (void)in_sizes; (void)n_in; (void)out_size; (void)ws_size;
    const float* x           = (const float*)d_in[0];
    const float* qkv_w       = (const float*)d_in[1];
    const float* qkv_b       = (const float*)d_in[2];
    const float* proj_w      = (const float*)d_in[3];
    const float* proj_b      = (const float*)d_in[4];
    const float* logit_scale = (const float*)d_in[5];
    const float* meta_w1     = (const float*)d_in[6];
    const float* meta_b1     = (const float*)d_in[7];
    const float* meta_w2     = (const float*)d_in[8];
    const float* meta_b2     = (const float*)d_in[9];
    const float* norm1_g     = (const float*)d_in[10];
    const float* norm1_b     = (const float*)d_in[11];
    const float* norm2_g     = (const float*)d_in[12];
    const float* norm2_b     = (const float*)d_in[13];
    const float* mlp_w1      = (const float*)d_in[14];
    const float* mlp_b1      = (const float*)d_in[15];
    const float* mlp_w2      = (const float*)d_in[16];
    const float* mlp_b2      = (const float*)d_in[17];

    char* ws = (char*)d_ws;
    unsigned short* wqkvT  = (unsigned short*)(ws + 0);          // 256x768 bf16
    unsigned short* wprojT = (unsigned short*)(ws + 393216);     // 256x256 bf16
    unsigned short* w1T    = (unsigned short*)(ws + 524288);     // 256x1024 bf16
    unsigned short* w2T    = (unsigned short*)(ws + 1048576);    // 1024x256 bf16
    float*          biasT  = (float*)(ws + 1572864);             // 8x64x64 f32
    float*          qf     = (float*)(ws + 1703936);             // 1024x8x64x32 f32
    float*          kf     = (float*)(ws + 1703936 + 67108864ull);
    float*          vf     = (float*)(ws + 1703936 + 2ull * 67108864ull);
    unsigned short* ctx    = (unsigned short*)(ws + 1703936 + 3ull * 67108864ull);

    float* xout = (float*)d_out;   // also serves as x1 between k_proj and k_mlp

    k_prep<<<3072, 256, 0, stream>>>(qkv_w, proj_w, mlp_w1, mlp_w2,
                                     wqkvT, wprojT, w1T, w2T);
    k_bias<<<16, 256, 0, stream>>>(meta_w1, meta_b1, meta_w2, meta_b2, biasT);
    k_qkv<<<1024, 256, 0, stream>>>(x, wqkvT, qkv_b, qf, kf, vf);
    k_attn<<<2048, 128, 0, stream>>>(qf, kf, vf, logit_scale, biasT, ctx);
    k_proj<<<1024, 256, 0, stream>>>(ctx, wprojT, proj_b, x, norm1_g, norm1_b, xout);
    k_mlp<<<1024, 256, 0, stream>>>(xout, w1T, mlp_b1, w2T, mlp_b2, norm2_g, norm2_b);
}
